// Recurrence_73632919322925
// MI455X (gfx1250) — matrix operand order repaired and checked by reading, not compiled, not run on hardware
//
#include <hip/hip_runtime.h>
#include <math.h>

// ---- sizes ----
#define HD     128     // H
#define NLN    32      // NL
#define NB     128     // N
#define TT_    64      // T
#define SOUT   212

typedef _Float16 half8_t __attribute__((ext_vector_type(8)));
typedef _Float16 v16h    __attribute__((ext_vector_type(16)));
typedef float    v8f     __attribute__((ext_vector_type(8)));

// ---------- WMMA fragment helpers ----------
// CDNA5 v_wmma_f32_16x16x32_f16 operand layouts (ISA 7.12.2):
//  A (16x32 f16): lane row = lane&15; element j -> K = (j>>3)*16 + (lane>>4)*8 + (j&7)
//    => per lane: two contiguous 8-half (16B) loads at k0+hi*8 and k0+16+hi*8
//  B (32x16 f16): lane col = lane&15; element j -> K = (j>>3)*16 + (lane>>4)*8 + (j&7)
//    => per lane: two contiguous 8-half loads at k0+hi*8 and k0+16+hi*8 from weight row [out][in]
//  C/D (16x16 f32): element r -> row r + (lane>>4)*8, col = lane&15
__device__ __forceinline__ v16h frag2(const _Float16* p0, const _Float16* p1) {
  half8_t a = *(const half8_t*)p0;
  half8_t b = *(const half8_t*)p1;
  v16h r;
#pragma unroll
  for (int j = 0; j < 8; ++j) { r[j] = a[j]; r[j + 8] = b[j]; }
  return r;
}
__device__ __forceinline__ v16h loadA(const _Float16* rowptr, int k0, int hi) {
  const _Float16* base = rowptr + k0 + hi * 8;
  return frag2(base, base + 16);
}
__device__ __forceinline__ v16h loadA_f32(const float* rowptr, int k0, int hi) {
  const float* base = rowptr + k0 + hi * 8;
  v16h r;
#pragma unroll
  for (int j = 0; j < 8; ++j) { r[j] = (_Float16)base[j]; r[j + 8] = (_Float16)base[16 + j]; }
  return r;
}
__device__ __forceinline__ v16h loadB(const _Float16* W, int ldk, int col, int k0, int hi) {
  const _Float16* base = W + (size_t)col * ldk + k0 + hi * 8;
  return frag2(base, base + 16);
}
__device__ __forceinline__ v8f wmma16(v16h a, v16h b, v8f c) {
  return __builtin_amdgcn_wmma_f32_16x16x32_f16(false, a, false, b, (short)0, c, false, false);
}
__device__ __forceinline__ float sigm(float x) { return 1.0f / (1.0f + expf(-x)); }

// ---------- small prep kernels ----------
__global__ void k_cvt_f16(_Float16* dst, const float* src, int n) {
  int i = blockIdx.x * blockDim.x + threadIdx.x;
  if (i < n) dst[i] = (_Float16)src[i];
}
__global__ void k_zero_f16(_Float16* dst, int n) {
  int i = blockIdx.x * blockDim.x + threadIdx.x;
  if (i < n) dst[i] = (_Float16)0.f;
}
// M_h[n][l][h] = (f16) embed[lines[n][l]][h]
__global__ void k_gather_M(_Float16* M_h, const float* __restrict__ inputs,
                           const float* __restrict__ embed, int total) {
  int i = blockIdx.x * blockDim.x + threadIdx.x;
  if (i >= total) return;
  int h = i & 127, l = (i >> 7) & 31, n = i >> 12;
  int idx = (int)inputs[(size_t)n * 98 + 64 + l];
  M_h[i] = (_Float16)embed[(size_t)idx * HD + h];
}

// ---------- encoder: bi-GRU over 4096 rows x 32 steps, fused mlp2 ----------
__global__ __launch_bounds__(256) void k_encoder(
    const _Float16* __restrict__ M_h,      // [128][32][128]
    const _Float16* __restrict__ Wih_h,    // [2][384][256]
    const _Float16* __restrict__ Whh_h,    // [2][384][128]
    const float* __restrict__ bih,         // [2][384]
    const float* __restrict__ bhh,         // [2][384]
    const _Float16* __restrict__ W1_h,     // [128][128]
    const float* __restrict__ b1,          // [128]
    const _Float16* __restrict__ W2_h,     // [16][128] (rows 8..15 zero)
    const float* __restrict__ b2,          // [8]
    float* __restrict__ Bbuf)              // [2][32][128][32][8]
{
  __shared__ __align__(16) float    s_gi[16][384];
  __shared__ __align__(16) float    s_gh[16][384];
  __shared__ __align__(16) float    s_h32[16][128];
  __shared__ __align__(16) _Float16 s_h16[16][128];
  __shared__ __align__(16) _Float16 s_z16[16][128];

  const int tid = threadIdx.x;
  const int wave = tid >> 5, lane = tid & 31, hi = lane >> 4, lm = lane & 15;
  const int dir = blockIdx.y;
  const int g0 = blockIdx.x * 16;        // global row base, rows g0..g0+15
  const int r = g0 >> 7;                 // roll index (uniform in block)
  const int nbase = g0 & 127;

  const _Float16* Wih = Wih_h + (size_t)dir * 384 * 256;
  const _Float16* Whh = Whh_h + (size_t)dir * 384 * 128;
  const float* bi = bih + dir * 384;
  const float* bh = bhh + dir * 384;

  for (int k = tid; k < 16 * 128; k += 256) {
    ((float*)s_h32)[k] = 0.f;
    ((_Float16*)s_h16)[k] = (_Float16)0.f;
  }
  __syncthreads();

  const int myn = nbase + lm;
  const _Float16* rowM2 = M_h + ((size_t)myn * 32 + r) * 128;   // "first" half of X
  const _Float16* rowH  = &s_h16[lm][0];

  for (int s = 0; s < 32; ++s) {
    const int jj = dir ? (31 - s) : s;                 // position in original sequence
    const int l1 = (r + jj) & 31;
    const _Float16* rowM1 = M_h + ((size_t)myn * 32 + l1) * 128;

    // gates: gi = x@Wih.T (K=256), gh = h@Whh.T (K=128); 24 col-tiles / 8 waves
#pragma unroll
    for (int tt = 0; tt < 3; ++tt) {
      const int ct = wave * 3 + tt;
      const int col = ct * 16 + lm;
      v8f ai = {}; v8f ah = {};
#pragma unroll
      for (int kc = 0; kc < 8; ++kc) {
        v16h a = (kc < 4) ? loadA(rowM1, kc * 32, hi) : loadA(rowM2, kc * 32 - 128, hi);
        v16h b = loadB(Wih, 256, col, kc * 32, hi);
        ai = wmma16(a, b, ai);
      }
#pragma unroll
      for (int kc = 0; kc < 4; ++kc) {
        v16h a = loadA(rowH, kc * 32, hi);
        v16h b = loadB(Whh, 128, col, kc * 32, hi);
        ah = wmma16(a, b, ah);
      }
#pragma unroll
      for (int rr = 0; rr < 8; ++rr) {
        const int row = rr + hi * 8;
        s_gi[row][col] = ai[rr];
        s_gh[row][col] = ah[rr];
      }
    }
    __syncthreads();

    // GRU gate math
#pragma unroll
    for (int k = 0; k < 8; ++k) {
      const int id = tid + k * 256;
      const int m = id >> 7, u = id & 127;
      float rg = sigm(s_gi[m][u] + bi[u] + s_gh[m][u] + bh[u]);
      float zg = sigm(s_gi[m][128 + u] + bi[128 + u] + s_gh[m][128 + u] + bh[128 + u]);
      float ng = tanhf(s_gi[m][256 + u] + bi[256 + u] + rg * (s_gh[m][256 + u] + bh[256 + u]));
      float hn = (1.f - zg) * ng + zg * s_h32[m][u];
      s_h32[m][u] = hn;
      s_h16[m][u] = (_Float16)hn;
    }
    __syncthreads();

    // fused mlp2 layer 1: h1 = relu(h @ W1.T + b1), one col-tile per wave
    {
      const int col = wave * 16 + lm;
      v8f acc = {};
#pragma unroll
      for (int kc = 0; kc < 4; ++kc) {
        v16h a = loadA(rowH, kc * 32, hi);
        v16h b = loadB(W1_h, 128, col, kc * 32, hi);
        acc = wmma16(a, b, acc);
      }
      const float bb = b1[col];
#pragma unroll
      for (int rr = 0; rr < 8; ++rr) {
        const int row = rr + hi * 8;
        float v = acc[rr] + bb;
        s_z16[row][col] = (_Float16)fmaxf(v, 0.f);
      }
    }
    __syncthreads();

    // fused mlp2 layer 2: B = sigmoid(h1 @ W2.T + b2), single 16x16 tile (8 valid cols)
    if (wave == 0) {
      v8f acc = {};
#pragma unroll
      for (int kc = 0; kc < 4; ++kc) {
        v16h a = loadA(&s_z16[lm][0], kc * 32, hi);
        v16h b = loadB(W2_h, 128, lm, kc * 32, hi);
        acc = wmma16(a, b, acc);
      }
      if (lm < 8) {
        const float bb = b2[lm];
#pragma unroll
        for (int rr = 0; rr < 8; ++rr) {
          const int row = rr + hi * 8;
          const int n = nbase + row;
          Bbuf[(((size_t)dir * 32 + r) * 128 + n) * 256 + jj * 8 + lm] = sigm(acc[rr] + bb);
        }
      }
    }
    __syncthreads();
  }
}

// ---------- stick-breaking: Bbuf -> P[32][128][64][16] (with pad/reverse/roll baked in) ----------
__global__ void k_stick(const float* __restrict__ Bbuf, float* __restrict__ P) {
  int id = blockIdx.x * blockDim.x + threadIdx.x;   // 65536 = 32*128*2*8
  int ee = id & 7;
  int d  = (id >> 3) & 1;
  int n  = (id >> 4) & 127;
  int rr = (id >> 11) & 31;
  const float* q = Bbuf + (((size_t)d * 32 + rr) * 128 + n) * 256;  // [j][8]
  const int rdst = (rr + 31) & 31;   // roll -1 on axis 0
  float c = 1.f;
  for (int tt = 0; tt < 64; ++tt) {
    float val;
    if (tt < 32) {
      val = 0.f;
    } else if (tt < 63) {
      int j = tt - 32;
      int jsrc = d ? (31 - j) : j;     // B1 backward slice is reversed
      float qv = q[jsrc * 8 + ee];
      val = qv * c;
      c *= (1.f - qv);
    } else {
      val = c;                          // forced terminal "1" times survival
    }
    int tdst, edst;
    if (d == 0) { tdst = (tt + 1) & 63; edst = ee + 8; }      // f goes to e>=8, roll +1 on t
    else        { tdst = (64 - tt) & 63; edst = ee; }          // b reversed then rolled
    P[(((size_t)rdst * 128 + n) * 64 + tdst) * 16 + edst] = val;
  }
}

// ---------- decoder: 64 sequential steps, row-independent, 8 blocks x 16 rows ----------
__global__ __launch_bounds__(256) void k_decoder(
    const float* __restrict__ inputs,     // [64][128][98]
    const float* __restrict__ rnn_hxs,    // [1][128][212]
    const _Float16* __restrict__ M_h,     // [128][32][128]
    const _Float16* __restrict__ Wih_h,   // [384][192]
    const _Float16* __restrict__ Whh_h,   // [384][128]
    const float* __restrict__ bih,        // [384]
    const float* __restrict__ bhh,        // [384]
    const _Float16* __restrict__ mlpW_h,  // [2][128][128]
    const float* __restrict__ mlp_b,      // [2][128]
    const _Float16* __restrict__ Whead_h, // [48][128]: actor16 | opt16 | critic1+pad
    const float* __restrict__ actor_b,    // [16]
    const float* __restrict__ opt_b,      // [16]
    const float* __restrict__ critic_b,   // [1]
    const float* __restrict__ P,          // [32][128][64][16]
    float* __restrict__ out)              // [64][128][212]
{
  __shared__ __align__(16) float    s_gi[16][384];
  __shared__ __align__(16) float    s_gh[16][384];
  __shared__ __align__(16) float    s_h32[16][128];
  __shared__ __align__(16) _Float16 s_h16[16][128];
  __shared__ __align__(16) _Float16 s_z1[16][128];
  __shared__ __align__(16) _Float16 s_z2[16][128];
  __shared__ float s_head[16][48];
  __shared__ float s_osm[16][16];
  __shared__ int   s_w[16], s_wold[16];

  const int tid = threadIdx.x;
  const int wave = tid >> 5, lane = tid & 31, hi = lane >> 4, lm = lane & 15;
  const int g0 = blockIdx.x * 16;

  for (int k = tid; k < 16 * 128; k += 256) {
    int m = k >> 7, u = k & 127;
    float hv = rnn_hxs[(size_t)(g0 + m) * 212 + 4 + u];
    s_h32[m][u] = hv;
    s_h16[m][u] = (_Float16)hv;
  }
  if (tid < 16) s_w[tid] = (int)rnn_hxs[(size_t)(g0 + tid) * 212 + 2];
  __syncthreads();

  const int myn = g0 + lm;
  const _Float16* rowH = &s_h16[lm][0];

  for (int t = 0; t < TT_; ++t) {
    const float* crow = inputs + ((size_t)t * 128 + myn) * 98;
    const _Float16* mrow = M_h + ((size_t)myn * 32 + s_w[lm]) * 128;   // x = [cond, M[n,w]]

    // GRU cell gates: K=192 input (2 cond chunks + 4 embed chunks) + K=128 hidden
#pragma unroll
    for (int tt = 0; tt < 3; ++tt) {
      const int ct = wave * 3 + tt;
      const int col = ct * 16 + lm;
      v8f ai = {}; v8f ah = {};
#pragma unroll
      for (int kc = 0; kc < 6; ++kc) {
        v16h a = (kc < 2) ? loadA_f32(crow, kc * 32, hi) : loadA(mrow, kc * 32 - 64, hi);
        v16h b = loadB(Wih_h, 192, col, kc * 32, hi);
        ai = wmma16(a, b, ai);
      }
#pragma unroll
      for (int kc = 0; kc < 4; ++kc) {
        v16h a = loadA(rowH, kc * 32, hi);
        v16h b = loadB(Whh_h, 128, col, kc * 32, hi);
        ah = wmma16(a, b, ah);
      }
#pragma unroll
      for (int rr = 0; rr < 8; ++rr) {
        const int row = rr + hi * 8;
        s_gi[row][col] = ai[rr];
        s_gh[row][col] = ah[rr];
      }
    }
    __syncthreads();

#pragma unroll
    for (int k = 0; k < 8; ++k) {
      const int id = tid + k * 256;
      const int m = id >> 7, u = id & 127;
      float rg = sigm(s_gi[m][u] + bih[u] + s_gh[m][u] + bhh[u]);
      float zg = sigm(s_gi[m][128 + u] + bih[128 + u] + s_gh[m][128 + u] + bhh[128 + u]);
      float ng = tanhf(s_gi[m][256 + u] + bih[256 + u] + rg * (s_gh[m][256 + u] + bhh[256 + u]));
      float hn = (1.f - zg) * ng + zg * s_h32[m][u];
      s_h32[m][u] = hn;
      s_h16[m][u] = (_Float16)hn;
      out[((size_t)t * 128 + (g0 + m)) * SOUT + 4 + u] = hn;
    }
    __syncthreads();

    // mlp layer 1
    {
      const int col = wave * 16 + lm;
      v8f acc = {};
#pragma unroll
      for (int kc = 0; kc < 4; ++kc) {
        v16h a = loadA(rowH, kc * 32, hi);
        v16h b = loadB(mlpW_h, 128, col, kc * 32, hi);
        acc = wmma16(a, b, acc);
      }
      const float bb = mlp_b[col];
#pragma unroll
      for (int rr = 0; rr < 8; ++rr) {
        const int row = rr + hi * 8;
        s_z1[row][col] = (_Float16)fmaxf(acc[rr] + bb, 0.f);
      }
    }
    __syncthreads();
    // mlp layer 2
    {
      const int col = wave * 16 + lm;
      v8f acc = {};
#pragma unroll
      for (int kc = 0; kc < 4; ++kc) {
        v16h a = loadA(&s_z1[lm][0], kc * 32, hi);
        v16h b = loadB(mlpW_h + 128 * 128, 128, col, kc * 32, hi);
        acc = wmma16(a, b, acc);
      }
      const float bb = mlp_b[128 + col];
#pragma unroll
      for (int rr = 0; rr < 8; ++rr) {
        const int row = rr + hi * 8;
        s_z2[row][col] = (_Float16)fmaxf(acc[rr] + bb, 0.f);
      }
    }
    __syncthreads();
    // heads: 3 col-tiles (actor / opt / critic+pad)
    if (wave < 3) {
      const int col = wave * 16 + lm;
      v8f acc = {};
#pragma unroll
      for (int kc = 0; kc < 4; ++kc) {
        v16h a = loadA(&s_z2[lm][0], kc * 32, hi);
        v16h b = loadB(Whead_h, 128, col, kc * 32, hi);
        acc = wmma16(a, b, acc);
      }
#pragma unroll
      for (int rr = 0; rr < 8; ++rr) s_head[rr + hi * 8][col] = acc[rr];
    }
    __syncthreads();

    // per-row scalar tail: softmaxes, value head, w update
    if (tid < 16) {
      const int m = tid, n = g0 + m;
      const size_t ob = ((size_t)t * 128 + n) * SOUT;
      float av[16], mx = -1e30f;
#pragma unroll
      for (int e = 0; e < 16; ++e) { av[e] = s_head[m][e] + actor_b[e]; mx = fmaxf(mx, av[e]); }
      float se = 0.f;
#pragma unroll
      for (int e = 0; e < 16; ++e) { av[e] = expf(av[e] - mx); se += av[e]; }
      float inv = 1.f / se;
#pragma unroll
      for (int e = 0; e < 16; ++e) out[ob + 132 + e] = av[e] * inv;

      float ov[16]; mx = -1e30f;
#pragma unroll
      for (int e = 0; e < 16; ++e) { ov[e] = s_head[m][16 + e] + opt_b[e]; mx = fmaxf(mx, ov[e]); }
      se = 0.f;
#pragma unroll
      for (int e = 0; e < 16; ++e) { ov[e] = expf(ov[e] - mx); se += ov[e]; }
      inv = 1.f / se;
#pragma unroll
      for (int e = 0; e < 16; ++e) s_osm[m][e] = ov[e] * inv;

      out[ob + 3] = s_head[m][32] + critic_b[0];
      float At = inputs[((size_t)t * 128 + n) * 98 + 96];
      float Wt = inputs[((size_t)t * 128 + n) * 98 + 97];
      out[ob + 0] = At;
      out[ob + 1] = Wt;
      s_wold[m] = s_w[m];
      int wn = s_w[m] + (int)Wt - 32;
      wn = wn < 0 ? 0 : (wn > 31 ? 31 : wn);
      s_w[m] = wn;
      out[ob + 2] = (float)wn;
    }
    __syncthreads();

    // p[n][ts] = sum_e P[w_old, n, ts, e] * o[n][e]
#pragma unroll
    for (int k = 0; k < 4; ++k) {
      const int id = tid + k * 256;
      const int m = id >> 6, ts = id & 63;
      const int n = g0 + m;
      const float* Pg = P + (((size_t)s_wold[m] * 128 + n) * 64 + ts) * 16;
      float acc = 0.f;
#pragma unroll
      for (int e = 0; e < 16; ++e) acc += Pg[e] * s_osm[m][e];
      out[((size_t)t * 128 + n) * SOUT + 148 + ts] = acc;
    }
    __syncthreads();
  }
}

// ---------- host orchestration ----------
extern "C" void kernel_launch(void* const* d_in, const int* in_sizes, int n_in,
                              void* d_out, int out_size, void* d_ws, size_t ws_size,
                              hipStream_t stream) {
  (void)in_sizes; (void)n_in; (void)out_size; (void)ws_size;
  const float* inputs   = (const float*)d_in[0];
  const float* rnn_hxs  = (const float*)d_in[1];
  const float* embed    = (const float*)d_in[2];
  const float* enc_Wih  = (const float*)d_in[3];
  const float* enc_Whh  = (const float*)d_in[4];
  const float* enc_bih  = (const float*)d_in[5];
  const float* enc_bhh  = (const float*)d_in[6];
  const float* cell_Wih = (const float*)d_in[7];
  const float* cell_Whh = (const float*)d_in[8];
  const float* cell_bih = (const float*)d_in[9];
  const float* cell_bhh = (const float*)d_in[10];
  const float* mlp_W    = (const float*)d_in[11];
  const float* mlp_b    = (const float*)d_in[12];
  const float* opt_W    = (const float*)d_in[13];
  const float* opt_b    = (const float*)d_in[14];
  const float* mlp2_W1  = (const float*)d_in[15];
  const float* mlp2_b1  = (const float*)d_in[16];
  const float* mlp2_W2  = (const float*)d_in[17];
  const float* mlp2_b2  = (const float*)d_in[18];
  const float* critic_W = (const float*)d_in[19];
  const float* critic_b = (const float*)d_in[20];
  const float* actor_W  = (const float*)d_in[21];
  const float* actor_b  = (const float*)d_in[22];
  float* out = (float*)d_out;

  char* ws = (char*)d_ws;
  size_t off = 0;
  auto alloc = [&](size_t bytes) -> void* {
    void* p = ws + off;
    off = (off + bytes + 255) & ~(size_t)255;
    return p;
  };
  _Float16* M_h       = (_Float16*)alloc((size_t)128 * 32 * 128 * 2);
  _Float16* encWih_h  = (_Float16*)alloc((size_t)2 * 384 * 256 * 2);
  _Float16* encWhh_h  = (_Float16*)alloc((size_t)2 * 384 * 128 * 2);
  _Float16* cellWih_h = (_Float16*)alloc((size_t)384 * 192 * 2);
  _Float16* cellWhh_h = (_Float16*)alloc((size_t)384 * 128 * 2);
  _Float16* mlpW_h    = (_Float16*)alloc((size_t)2 * 128 * 128 * 2);
  _Float16* mlp2W1_h  = (_Float16*)alloc((size_t)128 * 128 * 2);
  _Float16* mlp2W2p_h = (_Float16*)alloc((size_t)16 * 128 * 2);
  _Float16* heads_h   = (_Float16*)alloc((size_t)48 * 128 * 2);
  float*    Bbuf      = (float*)alloc((size_t)2 * 32 * 128 * 32 * 8 * 4);
  float*    Pmat      = (float*)alloc((size_t)32 * 128 * 64 * 16 * 4);

  auto cvt = [&](_Float16* d, const float* s, int n) {
    hipLaunchKernelGGL(k_cvt_f16, dim3((n + 255) / 256), dim3(256), 0, stream, d, s, n);
  };
  cvt(encWih_h, enc_Wih, 2 * 384 * 256);
  cvt(encWhh_h, enc_Whh, 2 * 384 * 128);
  cvt(cellWih_h, cell_Wih, 384 * 192);
  cvt(cellWhh_h, cell_Whh, 384 * 128);
  cvt(mlpW_h, mlp_W, 2 * 128 * 128);
  cvt(mlp2W1_h, mlp2_W1, 128 * 128);
  cvt(mlp2W2p_h, mlp2_W2, 8 * 128);
  hipLaunchKernelGGL(k_zero_f16, dim3(4), dim3(256), 0, stream, mlp2W2p_h + 8 * 128, 8 * 128);
  cvt(heads_h, actor_W, 16 * 128);
  cvt(heads_h + 16 * 128, opt_W, 16 * 128);
  cvt(heads_h + 32 * 128, critic_W, 128);
  hipLaunchKernelGGL(k_zero_f16, dim3(8), dim3(256), 0, stream, heads_h + 33 * 128, 15 * 128);

  hipLaunchKernelGGL(k_gather_M, dim3(2048), dim3(256), 0, stream, M_h, inputs, embed,
                     128 * 32 * 128);

  hipLaunchKernelGGL(k_encoder, dim3(256, 2), dim3(256), 0, stream,
                     M_h, encWih_h, encWhh_h, enc_bih, enc_bhh,
                     mlp2W1_h, mlp2_b1, mlp2W2p_h, mlp2_b2, Bbuf);

  hipLaunchKernelGGL(k_stick, dim3(256), dim3(256), 0, stream, Bbuf, Pmat);

  hipLaunchKernelGGL(k_decoder, dim3(8), dim3(256), 0, stream,
                     inputs, rnn_hxs, M_h, cellWih_h, cellWhh_h, cell_bih, cell_bhh,
                     mlpW_h, mlp_b, heads_h, actor_b, opt_b, critic_b, Pmat, out);
}
